// ClassicalAttentionLayer_41832981463178
// MI455X (gfx1250) — compile-verified, matrix operand order checked by reading, not run on hardware
//
#include <hip/hip_runtime.h>
#include <stdint.h>

// ---------------------------------------------------------------------------
// Fused attention for MI455X (gfx1250, wave32, WMMA bf16 16x16x32).
//   out = softmax( (xW_q)(xW_k)^T / 8 ) (xW_v)
// Pass 0: streaming f32->bf16 convert of x and w (hw v_cvt_pk_bf16_f32).
// Pass 1: qkv projection; one wave = one 16-row m-tile x one region (Q|K|V),
//         1536 waves, double-buffered frags -> staggered partial loadcnt waits.
// Pass 2: split-K flash attention, software-pipelined; 4096 waves.
// ---------------------------------------------------------------------------

typedef __attribute__((ext_vector_type(16))) __bf16 v16bf;
typedef __attribute__((ext_vector_type(2)))  __bf16 v2bf;
typedef __attribute__((ext_vector_type(8)))  float  v8f;

#define N_ROWS 8192
#define D_IN   512
#define D_HEAD 64

#if defined(__has_builtin)
#if __has_builtin(__builtin_amdgcn_cvt_pk_bf16_f32)
#define HAVE_CVT_PK_BF16 1
#endif
#endif
#ifndef HAVE_CVT_PK_BF16
#define HAVE_CVT_PK_BF16 0
#endif

static __device__ __forceinline__ unsigned short f2bf(float f) {
    union { float f; unsigned u; } x; x.f = f;
    unsigned r = x.u + 0x7FFFu + ((x.u >> 16) & 1u);   // round-to-nearest-even
    return (unsigned short)(r >> 16);
}

// pack two f32 -> two bf16 in one dword (one VALU op when hw cvt exists)
static __device__ __forceinline__ unsigned pk2bf(float a, float b) {
#if HAVE_CVT_PK_BF16
    union { v2bf v; unsigned u; } c;
    c.v = __builtin_amdgcn_cvt_pk_bf16_f32(a, b);
    return c.u;
#else
    return (unsigned)f2bf(a) | ((unsigned)f2bf(b) << 16);
#endif
}

union BF16x16 {               // one 16x16x32 bf16 WMMA operand (32 bytes)
    unsigned short s[16];
    uint4          u[2];
    v16bf          v;
};

static __device__ __forceinline__ v8f zero8() {
    v8f z;
#pragma unroll
    for (int i = 0; i < 8; ++i) z[i] = 0.0f;
    return z;
}

// ---------------------------------------------------------------------------
// Pass 0: streaming f32 -> bf16 (8 elements / thread, b128 in / b128 out)
// ---------------------------------------------------------------------------
__global__ __launch_bounds__(256)
void cvt_bf16_kernel(const float* __restrict__ src,
                     unsigned int* __restrict__ dst, int n8) {
    int i = blockIdx.x * blockDim.x + threadIdx.x;
    if (i >= n8) return;
    const float4* s = (const float4*)src + (size_t)i * 2;
    float4 f0 = s[0], f1 = s[1];
    uint4 o;
    o.x = pk2bf(f0.x, f0.y);
    o.y = pk2bf(f0.z, f0.w);
    o.z = pk2bf(f1.x, f1.y);
    o.w = pk2bf(f1.z, f1.w);
    ((uint4*)dst)[i] = o;
}

// ---------------------------------------------------------------------------
// Pass 1: qkv = x @ w^T (all bf16 in).
// One wave = one 16-row m-tile x one region (Q, K or V = 4 n-tiles).
//   1536 waves total; 4 accumulators + double-buffered {A, B[4]} frags
//   (~130 VGPRs) so both halves of the unroll-2 loop stay pipelined.
// A-frag of x and B-frag of w share the per-lane pattern:
//   lane holds row (lane%16), 16 contiguous k at offset (lane/16)*16.
// ---------------------------------------------------------------------------
__global__ __launch_bounds__(256)
void qkv_proj_kernel(const unsigned short* __restrict__ xbf,
                     const unsigned short* __restrict__ wbf,
                     unsigned short* __restrict__ qbf,
                     unsigned short* __restrict__ kbf,
                     unsigned short* __restrict__ vt) {
    const int lane = threadIdx.x & 31;
    const int wid  = blockIdx.x * (blockDim.x >> 5) + (threadIdx.x >> 5); // 0..1535
    const int r = lane & 15;
    const int h = lane >> 4;
    const int mt     = wid / 3;        // m-tile 0..511
    const int region = wid - mt * 3;   // 0=Q, 1=K, 2=V
    const int m0 = mt * 16;

    const unsigned short* xrow = xbf + (size_t)(m0 + r) * D_IN + h * 16;
    const unsigned short* wrow = wbf + (size_t)(region * 64 + r) * D_IN + h * 16;

    v8f acc[4];
#pragma unroll
    for (int i = 0; i < 4; ++i) acc[i] = zero8();

    BF16x16 aA, bA[4], aB, bB[4];

    auto load_frags = [&](BF16x16& a, BF16x16 (&b)[4], int k0) {
        const uint4* xp = (const uint4*)(xrow + k0);
        a.u[0] = xp[0];
        a.u[1] = xp[1];
#pragma unroll
        for (int i = 0; i < 4; ++i) {
            const uint4* wp = (const uint4*)(wrow + (size_t)i * 16 * D_IN + k0);
            b[i].u[0] = wp[0];
            b[i].u[1] = wp[1];
        }
    };
    auto mma4 = [&](const BF16x16& a, const BF16x16 (&b)[4]) {
#pragma unroll
        for (int i = 0; i < 4; ++i)
            acc[i] = __builtin_amdgcn_wmma_f32_16x16x32_bf16(
                false, a.v, false, b[i].v, (short)0, acc[i], false, false);
    };

    load_frags(aA, bA, 0);
#pragma unroll 1
    for (int k0 = 0; k0 < D_IN; k0 += 64) {
        load_frags(aB, bB, k0 + 32);       // prefetch odd step
        mma4(aA, bA);                      // covers odd-step load latency
        if (k0 + 64 < D_IN)
            load_frags(aA, bA, k0 + 64);   // prefetch next even step
        mma4(aB, bB);
    }

    // C/D layout: lane holds col n=lane%16, rows m = vgpr + 8*(lane/16).
    if (region < 2) {
        unsigned short* dst = (region == 0) ? qbf : kbf;
#pragma unroll
        for (int i = 0; i < 4; ++i) {
            const int n = i * 16 + r;        // column inside the 64-wide head
#pragma unroll
            for (int j = 0; j < 8; j += 2) {
                // hw packed convert, then split halves (rows differ by 1).
                const unsigned u = pk2bf(acc[i][j], acc[i][j + 1]);
                dst[(size_t)(m0 + j + 8 * h) * D_HEAD + n]     = (unsigned short)u;
                dst[(size_t)(m0 + j + 1 + 8 * h) * D_HEAD + n] = (unsigned short)(u >> 16);
            }
        }
    } else {
        // V transposed: lane's 8 values are at consecutive m -> one b128.
#pragma unroll
        for (int i = 0; i < 4; ++i) {
            const int n = i * 16 + r;
            uint4 o;
            o.x = pk2bf(acc[i][0], acc[i][1]);
            o.y = pk2bf(acc[i][2], acc[i][3]);
            o.z = pk2bf(acc[i][4], acc[i][5]);
            o.w = pk2bf(acc[i][6], acc[i][7]);
            *(uint4*)(vt + (size_t)n * N_ROWS + m0 + 8 * h) = o;
        }
    }
}

// ---------------------------------------------------------------------------
// Pass 2: split-K flash attention.
//   Workgroup (256 thr, 8 waves) owns queries [16*blockIdx.x, +16).
//   Wave w scans keys [1024*w, +1024) with online softmax:
//     S^T tile  = WMMA(A=K rows, B=Q rows)   -> lane owns query q=lane%16
//     O^T accum = WMMA(A=Vt rows, B=P^T)     -> lane owns query q=lane%16
//   P^T B-frag == P A-frag pattern, via a per-wave 1KB LDS round trip.
//   Pipelining: V-frags issued at block top; next K-frags issued right after
//   the QK WMMAs so they overlap the softmax VALU/trans work.
//   Partial (M, L, O^T) merged across the 8 waves in LDS (flash merge).
// ---------------------------------------------------------------------------
__global__ __launch_bounds__(256)
void flash_attn_kernel(const unsigned short* __restrict__ qbf,
                       const unsigned short* __restrict__ kbf,
                       const unsigned short* __restrict__ vt,
                       float* __restrict__ out) {
    __shared__ unsigned int plds[8][256];        // 1 KB/wave: 16x32 bf16 P tile
    __shared__ float sM[8][16];
    __shared__ float sL[8][16];
    __shared__ float sO[8][16][64];              // 32 KB partial O

    const int lane  = threadIdx.x & 31;
    const int wslot = threadIdx.x >> 5;
    const int r = lane & 15;
    const int h = lane >> 4;
    const int q0 = blockIdx.x * 16;
    unsigned int* pl = plds[wslot];
    const float scale = 0.125f;                  // 1/sqrt(64)

    // Q fragments (B-layout): lane = col q, d-chunk c covers d = 32c + h*16 .. +15
    BF16x16 qf[2];
#pragma unroll
    for (int c = 0; c < 2; ++c) {
        const uint4* qp = (const uint4*)(qbf + (size_t)(q0 + r) * D_HEAD + 32 * c + h * 16);
        qf[c].u[0] = qp[0];
        qf[c].u[1] = qp[1];
    }

    v8f ot[4];
#pragma unroll
    for (int t = 0; t < 4; ++t) ot[t] = zero8();
    float M = -INFINITY;
    float L = 0.0f;

    const int jbeg = wslot * (N_ROWS / 8);
    const int jend = jbeg + (N_ROWS / 8);

    BF16x16 kfA[2][2], kfB[2][2], vf[4];

    auto load_kf = [&](BF16x16 (&kf)[2][2], int j0) {
#pragma unroll
        for (int t = 0; t < 2; ++t)
#pragma unroll
            for (int c = 0; c < 2; ++c) {
                const uint4* kp = (const uint4*)(kbf + (size_t)(j0 + t * 16 + r) * D_HEAD
                                                 + 32 * c + h * 16);
                kf[t][c].u[0] = kp[0];
                kf[t][c].u[1] = kp[1];
            }
    };
    auto load_vf = [&](int j0) {
#pragma unroll
        for (int t = 0; t < 4; ++t) {
            const uint4* vp = (const uint4*)(vt + (size_t)(t * 16 + r) * N_ROWS
                                             + j0 + h * 16);
            vf[t].u[0] = vp[0];
            vf[t].u[1] = vp[1];
        }
    };

    // Online softmax + LDS transpose of P + PV WMMAs for one 32-key block.
    auto block_tail = [&](v8f s0, v8f s1) {
        float mx = -INFINITY;
#pragma unroll
        for (int j = 0; j < 8; ++j) mx = fmaxf(mx, s0[j]);
#pragma unroll
        for (int j = 0; j < 8; ++j) mx = fmaxf(mx, s1[j]);
        mx = fmaxf(mx, __shfl_xor(mx, 16));
        const float Mn    = fmaxf(M, mx);
        const float alpha = __expf(scale * (M - Mn));

        float p0[8], p1[8];
        float ls = 0.0f;
#pragma unroll
        for (int j = 0; j < 8; ++j) { p0[j] = __expf(scale * (s0[j] - Mn)); ls += p0[j]; }
#pragma unroll
        for (int j = 0; j < 8; ++j) { p1[j] = __expf(scale * (s1[j] - Mn)); ls += p1[j]; }
        ls += __shfl_xor(ls, 16);
        L = L * alpha + ls;
        M = Mn;
#pragma unroll
        for (int t = 0; t < 4; ++t) ot[t] = ot[t] * alpha;

        // P (16q x 32j) to LDS row-major bf16; read back as B-frag of P^T.
#pragma unroll
        for (int j = 0; j < 4; ++j) {
            pl[r * 16 + 4 * h + j]     = pk2bf(p0[2 * j], p0[2 * j + 1]);
            pl[r * 16 + 4 * h + 8 + j] = pk2bf(p1[2 * j], p1[2 * j + 1]);
        }
        BF16x16 pb;
        const uint4* pp = (const uint4*)(pl + r * 16 + h * 8);
        pb.u[0] = pp[0];
        pb.u[1] = pp[1];

#pragma unroll
        for (int t = 0; t < 4; ++t)
            ot[t] = __builtin_amdgcn_wmma_f32_16x16x32_bf16(
                false, vf[t].v, false, pb.v, (short)0, ot[t], false, false);
    };

    auto qk_mma = [&](const BF16x16 (&kf)[2][2], v8f& s0, v8f& s1) {
        s0 = zero8(); s1 = zero8();
        s0 = __builtin_amdgcn_wmma_f32_16x16x32_bf16(false, kf[0][0].v, false, qf[0].v, (short)0, s0, false, false);
        s0 = __builtin_amdgcn_wmma_f32_16x16x32_bf16(false, kf[0][1].v, false, qf[1].v, (short)0, s0, false, false);
        s1 = __builtin_amdgcn_wmma_f32_16x16x32_bf16(false, kf[1][0].v, false, qf[0].v, (short)0, s1, false, false);
        s1 = __builtin_amdgcn_wmma_f32_16x16x32_bf16(false, kf[1][1].v, false, qf[1].v, (short)0, s1, false, false);
    };

    load_kf(kfA, jbeg);
#pragma unroll 1
    for (int j0 = jbeg; j0 < jend; j0 += 64) {
        v8f s0, s1;
        // ---- block j0 (kfA resident) ----
        load_vf(j0);                       // V loads cover whole block latency
        qk_mma(kfA, s0, s1);
        load_kf(kfB, j0 + 32);             // overlaps softmax + LDS + PV
        block_tail(s0, s1);
        // ---- block j0+32 (kfB resident) ----
        load_vf(j0 + 32);
        qk_mma(kfB, s0, s1);
        if (j0 + 64 < jend)
            load_kf(kfA, j0 + 64);
        block_tail(s0, s1);
    }

    // --- stage partials: lane owns q = r, d = t*16 + vgpr + 8h -------------
    if (h == 0) {
        sM[wslot][r] = M;
        sL[wslot][r] = L;
    }
#pragma unroll
    for (int t = 0; t < 4; ++t)
#pragma unroll
        for (int j = 0; j < 8; ++j)
            sO[wslot][r][t * 16 + j + 8 * h] = ot[t][j];
    __syncthreads();

    // --- flash merge of the 8 partial states -------------------------------
    // thread -> (q = tid&15, d = 4*(tid>>4) .. +3); 256 threads cover 16x64.
    {
        const int q  = threadIdx.x & 15;
        const int d0 = (threadIdx.x >> 4) * 4;
        float Mg = sM[0][q];
#pragma unroll
        for (int w = 1; w < 8; ++w) Mg = fmaxf(Mg, sM[w][q]);
        float Lg = 0.0f;
        float acc[4] = {0.0f, 0.0f, 0.0f, 0.0f};
#pragma unroll
        for (int w = 0; w < 8; ++w) {
            const float wgt = __expf(scale * (sM[w][q] - Mg));
            Lg += wgt * sL[w][q];
#pragma unroll
            for (int i = 0; i < 4; ++i) acc[i] += wgt * sO[w][q][d0 + i];
        }
        const float inv = 1.0f / Lg;
        float4 o = { acc[0] * inv, acc[1] * inv, acc[2] * inv, acc[3] * inv };
        *(float4*)(out + (size_t)(q0 + q) * D_HEAD + d0) = o;
    }
}

// ---------------------------------------------------------------------------
extern "C" void kernel_launch(void* const* d_in, const int* in_sizes, int n_in,
                              void* d_out, int out_size, void* d_ws, size_t ws_size,
                              hipStream_t stream) {
    const float* x = (const float*)d_in[0];   // [8192, 512]
    const float* w = (const float*)d_in[1];   // [192, 512]
    float* out = (float*)d_out;               // [8192, 64]

    unsigned short* xbf = (unsigned short*)d_ws;                 // 8 MB
    unsigned short* wbf = xbf + (size_t)N_ROWS * D_IN;           // 192 KB
    unsigned short* qbf = wbf + (size_t)192 * D_IN;              // 1 MB
    unsigned short* kbf = qbf + (size_t)N_ROWS * D_HEAD;         // 1 MB
    unsigned short* vt  = kbf + (size_t)N_ROWS * D_HEAD;         // 1 MB (transposed)

    // Pass 0: convert x (4M elems) and w (96K elems) to bf16.
    cvt_bf16_kernel<<<(N_ROWS * D_IN / 8 + 255) / 256, 256, 0, stream>>>(
        x, (unsigned int*)xbf, N_ROWS * D_IN / 8);
    cvt_bf16_kernel<<<(192 * D_IN / 8 + 255) / 256, 256, 0, stream>>>(
        w, (unsigned int*)wbf, 192 * D_IN / 8);

    // Pass 1: 1536 waves (512 m-tiles x 3 regions), 8 waves / block.
    qkv_proj_kernel<<<192, 256, 0, stream>>>(xbf, wbf, qbf, kbf, vt);

    // Pass 2: one block per 16 queries; 8 waves split the key range.
    flash_attn_kernel<<<N_ROWS / 16, 256, 0, stream>>>(qbf, kbf, vt, out);
}
